// HierarchicalPooling_22179211116918
// MI455X (gfx1250) — matrix operand order repaired and checked by reading, not compiled, not run on hardware
//
#include <hip/hip_runtime.h>
#include <hip/hip_bf16.h>
#include <math.h>

// ---------------------------------------------------------------------------
// HierarchicalPooling on MI455X (gfx1250, wave32)
// B=8 S=4096 H=1024 E=256 K=64 Hq=256
// ---------------------------------------------------------------------------

typedef __attribute__((ext_vector_type(16))) __bf16          v16bf;
typedef __attribute__((ext_vector_type(8)))  float           v8f;
typedef __attribute__((ext_vector_type(4)))  unsigned int    v4u;
typedef __attribute__((ext_vector_type(4)))  unsigned short  v4s;

union BF16x16 {
    v16bf v;
    unsigned short u[16];
    v4u q[2];           // two 16-byte halves (elements 0-7, 8-15)
};

// round-to-nearest-even fp32 -> bf16 (bit-level, no libcalls)
static __device__ __forceinline__ unsigned short f2bf(float f) {
    unsigned int x = __float_as_uint(f);
    unsigned int r = x + 0x7FFFu + ((x >> 16) & 1u);
    return (unsigned short)(r >> 16);
}
static __device__ __forceinline__ float bf2f(unsigned short h) {
    return __uint_as_float(((unsigned int)h) << 16);
}

#define NEGBIG (-1e30f)

// ---------------------------------------------------------------------------
// Kernel 0: one-time fp32 -> bf16 hi/lo split of a weight matrix (n4 float4s)
// ---------------------------------------------------------------------------
__global__ __launch_bounds__(256)
void convert_hilo_kernel(const float* __restrict__ W,
                         unsigned short* __restrict__ Whi,
                         unsigned short* __restrict__ Wlo,
                         int n4)
{
    int i = blockIdx.x * 256 + threadIdx.x;
    if (i >= n4) return;
    float4 xv = ((const float4*)W)[i];
    v4s h, l;
    float f[4] = {xv.x, xv.y, xv.z, xv.w};
    #pragma unroll
    for (int e = 0; e < 4; ++e) {
        unsigned short hb = f2bf(f[e]);
        h[e] = hb;
        l[e] = f2bf(f[e] - bf2f(hb));
    }
    *(v4s*)(Whi + (size_t)i * 4) = h;
    *(v4s*)(Wlo + (size_t)i * 4) = l;
}

// ---------------------------------------------------------------------------
// Kernel 1/3: fused  scores = tanh(X @ W1 + b1) @ w2 + b2
//   X: [nrows, 1024] fp32 row-major; W1hi/W1lo: [1024, 256] bf16 (pre-split)
//   Workgroup = 8 waves = 256 thr, computes 128 rows x 256 cols.
//   Wave w owns N-tiles 2w, 2w+1 and loops 8 M-tiles.
//   K-chunk of 32 staged into LDS as bf16 hi/lo ONCE per workgroup, so the
//   inner loop is pure ds_load_b128 -> v_wmma (hi*hi + lo*hi + hi*lo).
// ---------------------------------------------------------------------------
__global__ __launch_bounds__(256)
void mlp_score_kernel(const float* __restrict__ X,
                      const unsigned short* __restrict__ W1hi,
                      const unsigned short* __restrict__ W1lo,
                      const float* __restrict__ b1,
                      const float* __restrict__ w2,
                      const float* __restrict__ b2,
                      float* __restrict__ scores,
                      int nrows)
{
    __shared__ unsigned short xh[128 * 32];   // 8 KB: A chunk, bf16 hi
    __shared__ unsigned short xl[128 * 32];   // 8 KB: A chunk, bf16 lo
    __shared__ float partial[8][128];         // 4 KB: per-wave score partials

    const int tid  = threadIdx.x;
    const int wave = tid >> 5;
    const int lane = tid & 31;
    const int l15  = lane & 15;
    const int sel  = lane >> 4;           // 0: lanes 0-15, 1: lanes 16-31
    const int m0   = blockIdx.x * 128;    // nrows is a multiple of 128

    v8f acc[8][2];
    {
        v8f z = {0.f, 0.f, 0.f, 0.f, 0.f, 0.f, 0.f, 0.f};
        for (int mt = 0; mt < 8; ++mt)
            for (int j = 0; j < 2; ++j)
                acc[mt][j] = z;
    }

    for (int kc = 0; kc < 1024; kc += 32) {
        __syncthreads();   // LDS reuse barrier
        // cooperative stage + hi/lo split: 128x32 floats = 1024 float4,
        // 4 float4 per thread, converted once per workgroup.
        for (int i = 0; i < 4; ++i) {
            int f4  = tid + i * 256;      // 0..1023
            int row = f4 >> 3;            // 8 float4 per row
            int c4  = f4 & 7;
            float4 xv =
                ((const float4*)(X + (size_t)(m0 + row) * 1024 + kc))[c4];
            float f[4] = {xv.x, xv.y, xv.z, xv.w};
            v4s h, l;
            #pragma unroll
            for (int e = 0; e < 4; ++e) {
                unsigned short hb = f2bf(f[e]);
                h[e] = hb;
                l[e] = f2bf(f[e] - bf2f(hb));
            }
            *(v4s*)(xh + f4 * 4) = h;
            *(v4s*)(xl + f4 * 4) = l;
        }
        __syncthreads();

        // --- B fragments for this wave's two N-tiles (pre-split bf16) -----
        // B layout (16b, 32x16): lane n = l15, elems 0-7 -> K = sel*8+0..7 ; elems 8-15 -> K = 16+sel*8+0..7
        BF16x16 bhi[2], blo[2];
        for (int j = 0; j < 2; ++j) {
            const int ncol = (wave * 2 + j) * 16 + l15;
            #pragma unroll
            for (int s = 0; s < 16; ++s) {
                const int kk = kc + ((s < 8) ? (sel * 8 + s) : (16 + sel * 8 + s - 8));
                bhi[j].u[s] = W1hi[(size_t)kk * 256 + ncol];
                blo[j].u[s] = W1lo[(size_t)kk * 256 + ncol];
            }
        }

        for (int mt = 0; mt < 8; ++mt) {
            // --- A fragment straight from LDS (already bf16) --------------
            // A layout (16b, 16x32): lane m = l15 (both halves),
            // elems 0-7 -> K = sel*8+0..7 ; elems 8-15 -> K = 16+sel*8+0..7
            const unsigned short* aph = xh + (mt * 16 + l15) * 32 + sel * 8;
            const unsigned short* apl = xl + (mt * 16 + l15) * 32 + sel * 8;
            BF16x16 ahi, alo;
            ahi.q[0] = *(const v4u*)(aph);
            ahi.q[1] = *(const v4u*)(aph + 16);
            alo.q[0] = *(const v4u*)(apl);
            alo.q[1] = *(const v4u*)(apl + 16);
            // hi/lo split: hi*hi + lo*hi + hi*lo  (f32 accumulate)
            for (int j = 0; j < 2; ++j) {
                acc[mt][j] = __builtin_amdgcn_wmma_f32_16x16x32_bf16(
                    false, ahi.v, false, bhi[j].v, (short)0, acc[mt][j], false, false);
                acc[mt][j] = __builtin_amdgcn_wmma_f32_16x16x32_bf16(
                    false, alo.v, false, bhi[j].v, (short)0, acc[mt][j], false, false);
                acc[mt][j] = __builtin_amdgcn_wmma_f32_16x16x32_bf16(
                    false, ahi.v, false, blo[j].v, (short)0, acc[mt][j], false, false);
            }
        }
    }

    // --- epilogue: tanh(+b1), dot with w2, reduce over N ------------------
    // C/D layout: VGPR r, lane l -> M = mt*16 + r + 8*sel, N = ntile*16 + l15
    for (int mt = 0; mt < 8; ++mt) {
        float contrib[8] = {0.f, 0.f, 0.f, 0.f, 0.f, 0.f, 0.f, 0.f};
        for (int j = 0; j < 2; ++j) {
            const int n = (wave * 2 + j) * 16 + l15;
            const float b1v = b1[n];
            const float w2v = w2[n];
            #pragma unroll
            for (int r = 0; r < 8; ++r)
                contrib[r] += tanhf(acc[mt][j][r] + b1v) * w2v;
        }
        // reduce across the 16 lanes that share the same row set
        #pragma unroll
        for (int msk = 1; msk < 16; msk <<= 1)
            #pragma unroll
            for (int r = 0; r < 8; ++r)
                contrib[r] += __shfl_xor(contrib[r], msk, 16);
        if (l15 == 0) {
            #pragma unroll
            for (int r = 0; r < 8; ++r)
                partial[wave][mt * 16 + sel * 8 + r] = contrib[r];
        }
    }
    __syncthreads();
    if (tid < 128) {
        float s = 0.f;
        for (int w = 0; w < 8; ++w) s += partial[w][tid];
        scores[m0 + tid] = s + b2[0];
    }
}

// ---------------------------------------------------------------------------
// Kernel 2: per-(b,k) segment masked softmax over tokens + weighted pooling.
//   Block k owns contiguous tokens [bnd[k], bnd[k+1]) (empty if duplicate).
//   One 256-thread workgroup per (b,k); thread t owns float4 column t.
// ---------------------------------------------------------------------------
__global__ __launch_bounds__(256)
void block_pool_kernel(const float* __restrict__ X,
                       const int*   __restrict__ amask,
                       const int*   __restrict__ bnd,
                       const float* __restrict__ tokScores,
                       float* __restrict__ blockEmbed,
                       float* __restrict__ blockDenom,
                       int Bv, int Sv, int Kv)
{
    const int bk = blockIdx.x;
    const int b  = bk / Kv;
    const int k  = bk % Kv;
    const int start = bnd[b * Kv + k];
    const int end   = (k == Kv - 1) ? Sv : bnd[b * Kv + k + 1];

    __shared__ float red[256];
    __shared__ float wbuf[256];
    const int tid = threadIdx.x;

    const float* sc = tokScores + (size_t)b * Sv;
    const int*   mk = amask     + (size_t)b * Sv;

    // pass 1: segment max (masked)
    float mx = -3.4e38f;
    for (int s = start + tid; s < end; s += 256)
        if (mk[s] == 1) mx = fmaxf(mx, sc[s]);
    red[tid] = mx;
    __syncthreads();
    for (int off = 128; off > 0; off >>= 1) {
        if (tid < off) red[tid] = fmaxf(red[tid], red[tid + off]);
        __syncthreads();
    }
    mx = red[0];
    __syncthreads();

    // pass 2: exp weights (staged per 256-token chunk) + weighted column sums
    float4 acc = {0.f, 0.f, 0.f, 0.f};
    float  dsum = 0.f;
    const float4* X4 = (const float4*)(X + (size_t)b * Sv * 1024);
    for (int base = start; base < end; base += 256) {
        const int t = base + tid;
        float w = 0.f;
        if (t < end && mk[t] == 1) w = __expf(sc[t] - mx);
        wbuf[tid] = w;
        dsum += w;
        __syncthreads();
        const int lim = min(256, end - base);
        for (int i = 0; i < lim; ++i) {
            if (i + 8 < lim)
                __builtin_prefetch(&X4[(size_t)(base + i + 8) * 256 + tid], 0, 1);
            const float wv = wbuf[i];
            if (wv > 0.f) {
                float4 xv = X4[(size_t)(base + i) * 256 + tid];
                acc.x += wv * xv.x; acc.y += wv * xv.y;
                acc.z += wv * xv.z; acc.w += wv * xv.w;
            }
        }
        __syncthreads();
    }

    red[tid] = dsum;
    __syncthreads();
    for (int off = 128; off > 0; off >>= 1) {
        if (tid < off) red[tid] += red[tid + off];
        __syncthreads();
    }
    const float denom = red[0];
    const float inv = (denom > 0.f) ? (1.f / fmaxf(denom, 1e-30f)) : 0.f;
    float4 outv = {acc.x * inv, acc.y * inv, acc.z * inv, acc.w * inv};
    ((float4*)(blockEmbed + (size_t)bk * 1024))[tid] = outv;
    if (tid == 0) blockDenom[bk] = denom;
}

// ---------------------------------------------------------------------------
// Kernel 4: per-batch block softmax, func = sum_k bw*embed, out = func@out_w
//           + out_b, then L2 normalize. One 256-thread workgroup per b.
// ---------------------------------------------------------------------------
__global__ __launch_bounds__(256)
void finalize_kernel(const float* __restrict__ blockEmbed,
                     const float* __restrict__ blockDenom,
                     const float* __restrict__ blkScores,
                     const float* __restrict__ out_w,
                     const float* __restrict__ out_b,
                     float* __restrict__ out,
                     int Kv)
{
    const int b   = blockIdx.x;
    const int tid = threadIdx.x;
    __shared__ float bw[64];
    __shared__ float funcLds[1024];
    __shared__ float red[256];

    if (tid == 0) {
        float s[64];
        float mxv = -3.4e38f;
        for (int k = 0; k < Kv; ++k) {
            float v = blkScores[b * Kv + k];
            v = (blockDenom[b * Kv + k] > 0.f) ? v : NEGBIG;
            s[k] = v;
            mxv = fmaxf(mxv, v);
        }
        float sum = 0.f;
        for (int k = 0; k < Kv; ++k) {
            float e = __expf(s[k] - mxv);
            bw[k] = e;
            sum += e;
        }
        const float inv = 1.f / sum;
        for (int k = 0; k < Kv; ++k) bw[k] *= inv;
    }
    __syncthreads();

    float4 f4 = {0.f, 0.f, 0.f, 0.f};
    const float4* BE4 = (const float4*)(blockEmbed + (size_t)b * Kv * 1024);
    for (int k = 0; k < Kv; ++k) {
        const float w = bw[k];
        float4 xv = BE4[(size_t)k * 256 + tid];
        f4.x += w * xv.x; f4.y += w * xv.y;
        f4.z += w * xv.z; f4.w += w * xv.w;
    }
    ((float4*)funcLds)[tid] = f4;
    __syncthreads();

    float accv = out_b[tid];
    for (int h = 0; h < 1024; ++h)
        accv += funcLds[h] * out_w[(size_t)h * 256 + tid];

    red[tid] = accv * accv;
    __syncthreads();
    for (int off = 128; off > 0; off >>= 1) {
        if (tid < off) red[tid] += red[tid + off];
        __syncthreads();
    }
    const float norm  = sqrtf(red[0]);
    const float scale = 1.f / fmaxf(norm, 1e-12f);
    out[b * 256 + tid] = accv * scale;
}

// ---------------------------------------------------------------------------
extern "C" void kernel_launch(void* const* d_in, const int* in_sizes, int n_in,
                              void* d_out, int out_size, void* d_ws, size_t ws_size,
                              hipStream_t stream)
{
    (void)in_sizes; (void)n_in; (void)out_size; (void)ws_size;
    const int B = 8, S = 4096, K = 64;

    const float* hidden   = (const float*)d_in[0];
    const int*   amask    = (const int*)  d_in[1];
    const int*   bnd      = (const int*)  d_in[2];
    const float* tok_w1   = (const float*)d_in[3];
    const float* tok_b1   = (const float*)d_in[4];
    const float* tok_w2   = (const float*)d_in[5];
    const float* tok_b2   = (const float*)d_in[6];
    const float* blk_w1   = (const float*)d_in[7];
    const float* blk_b1   = (const float*)d_in[8];
    const float* blk_w2   = (const float*)d_in[9];
    const float* blk_b2   = (const float*)d_in[10];
    const float* out_w    = (const float*)d_in[11];
    const float* out_b    = (const float*)d_in[12];
    float* out = (float*)d_out;

    // workspace layout
    char* ws = (char*)d_ws;
    float*          tokScores  = (float*)(ws);                      // 131072 B
    float*          blockEmbed = (float*)(ws + 131072);             // 2097152 B
    float*          blockDenom = (float*)(ws + 2228224);            // 2048 B
    float*          blkScores  = (float*)(ws + 2230272);            // 2048 B
    unsigned short* tokW1hi    = (unsigned short*)(ws + 2232320);   // 524288 B
    unsigned short* tokW1lo    = (unsigned short*)(ws + 2756608);   // 524288 B
    unsigned short* blkW1hi    = (unsigned short*)(ws + 3280896);   // 524288 B
    unsigned short* blkW1lo    = (unsigned short*)(ws + 3805184);   // 524288 B

    // 0) split W1 weights into bf16 hi/lo once (L2-resident afterwards)
    const int n4 = 1024 * 256 / 4;
    convert_hilo_kernel<<<n4 / 256, 256, 0, stream>>>(tok_w1, tokW1hi, tokW1lo, n4);
    convert_hilo_kernel<<<n4 / 256, 256, 0, stream>>>(blk_w1, blkW1hi, blkW1lo, n4);

    // 1) token scores: 32768 rows -> 256 workgroups
    mlp_score_kernel<<<256, 256, 0, stream>>>(
        hidden, tokW1hi, tokW1lo, tok_b1, tok_w2, tok_b2, tokScores, B * S);

    // 2) per-block masked softmax + pooling: one WG per (b,k)
    block_pool_kernel<<<B * K, 256, 0, stream>>>(
        hidden, amask, bnd, tokScores, blockEmbed, blockDenom, B, S, K);

    // 3) block scores: 512 rows -> 4 workgroups (same WMMA kernel)
    mlp_score_kernel<<<4, 256, 0, stream>>>(
        blockEmbed, blkW1hi, blkW1lo, blk_b1, blk_w2, blk_b2, blkScores, B * K);

    // 4) block softmax + projection + L2 normalize: one WG per b
    finalize_kernel<<<B, 256, 0, stream>>>(
        blockEmbed, blockDenom, blkScores, out_w, out_b, out, K);
}
